// GNN_29832842838644
// MI455X (gfx1250) — matrix operand order repaired and checked by reading, not compiled, not run on hardware
//
#include <hip/hip_runtime.h>

// ---------------------------------------------------------------------------
// Problem constants (match reference)
// ---------------------------------------------------------------------------
#define NNODE   16384
#define NHYD    4096
#define DIM     1024
#define KDIM    2048          // 2*DIM
#define NEDGE   65536
#define EHH     8192
#define EOO     40960

typedef __bf16 v16bf __attribute__((ext_vector_type(16)));
typedef __bf16 v8bf  __attribute__((ext_vector_type(8)));
typedef float  v8f   __attribute__((ext_vector_type(8)));
typedef unsigned v4u __attribute__((ext_vector_type(4)));
typedef int      v8i __attribute__((ext_vector_type(8)));
typedef int      v4i __attribute__((ext_vector_type(4)));

// Monotone float <-> unsigned key for atomicMax-based segment max
static __device__ __forceinline__ unsigned ordkey(float f) {
  unsigned b = __float_as_uint(f);
  return (b & 0x80000000u) ? ~b : (b | 0x80000000u);
}
static __device__ __forceinline__ float orddec(unsigned k) {
  return (k & 0x80000000u) ? __uint_as_float(k ^ 0x80000000u)
                           : __uint_as_float(~k);
}

// ---------------------------------------------------------------------------
// B-panel staging: move one 32KB k-panel (32 rows x 1KB, row stride 64KB in
// global) into contiguous LDS.
//   Preferred: Tensor Data Mover (one instruction, TENSORcnt).
//   Fallback:  per-lane global_load_async_to_lds_b128 (ASYNCcnt).
// ---------------------------------------------------------------------------
#if __has_builtin(__builtin_amdgcn_tensor_load_to_lds)
#define USE_TDM 1
#else
#define USE_TDM 0
#endif

static __device__ __forceinline__ void stage_panel(const __bf16* Wcb, int ks,
                                                   unsigned lds_byte,
                                                   int tid, int wave) {
#if USE_TDM
  if (wave == 0) {
    // D# group0: count=1 | lds_addr | global_addr(57b) | type=2
    unsigned long long ga =
        (unsigned long long)(uintptr_t)(Wcb + ((size_t)ks << 9));  // +ks*1KB
    v4u g0;
    g0[0] = 1u;                                     // count=1 (user, valid)
    g0[1] = lds_byte;                               // LDS byte address
    g0[2] = (unsigned)(ga & 0xFFFFFFFFu);           // global_addr[31:0]
    g0[3] = (unsigned)((ga >> 32) & 0x1FFFFFFu)     // global_addr[56:32]
            | (2u << 30);                           // type=2 ("image")
    // D# group1: data_size=4B; tensor 256x32; tile 256x32; dim0_stride=16384
    v8i g1;
    g1[0] = 0x00020000;            // wg_mask=0, data_size=2 (4B)
    g1[1] = (int)(256u << 16);     // tensor_dim0[15:0]  @ bits[63:48]
    g1[2] = (int)(32u << 16);      // tensor_dim1[15:0]  @ bits[95:80]
    g1[3] = (int)(256u << 16);     // tile_dim0 = 256    @ bits[127:112]
    g1[4] = 32;                    // tile_dim1 = 32     @ bits[143:128]
    g1[5] = 16384;                 // tensor_dim0_stride = 16384 (x4B = 64KB)
    g1[6] = 0;
    g1[7] = 0;
    v4i z4 = {0, 0, 0, 0};
    v8i z8 = {0, 0, 0, 0, 0, 0, 0, 0};
    // 6-arg form (clang-23 / therock headers): (g0, g1, g2, g3, g4, cpol)
    __builtin_amdgcn_tensor_load_to_lds(g0, g1, z4, z4, z8, 0);
  }
#else
  // 2048 x 16B chunks, grid-stride over the block
  for (int idx = tid; idx < 2048; idx += (int)blockDim.x) {
    int nt  = idx >> 6;
    int o4  = idx & 63;
    const __bf16* gp = Wcb + (((size_t)(nt * 64 + ks)) << 9) + (o4 << 3);
    unsigned lo = lds_byte + (unsigned)idx * 16u;
    asm volatile("global_load_async_to_lds_b128 %0, %1, off"
                 :: "v"(lo), "v"(gp) : "memory");
  }
#endif
}

static __device__ __forceinline__ void stage_wait() {
#if USE_TDM
  __builtin_amdgcn_s_wait_tensorcnt(0);
#else
  asm volatile("s_wait_asynccnt 0x0" ::: "memory");
#endif
}

// ---------------------------------------------------------------------------
// Prep: n_f (f32) -> bf16 into A_node[:, 0:1024]   (A_node is [N][2048] bf16)
// ---------------------------------------------------------------------------
__global__ void prep_nf_k(const float* __restrict__ nf,
                          __bf16* __restrict__ Anode) {
  size_t i = (size_t)blockIdx.x * blockDim.x + threadIdx.x;  // < N*D
  size_t row = i >> 10;
  size_t col = i & 1023;
  Anode[row * KDIM + col] = (__bf16)nf[i];
}

// ---------------------------------------------------------------------------
// Prep: pack 5 weight matrices [2048,1024] f32 into per-lane WMMA B-fragment
// order, bf16.  Tile t = nt*64 + kt, 512 bf16/tile, element r = lane*16 + i:
// B(k = kt*32 + 16*(lane>>4) + i, n = nt*16 + (lane&15)).
// ---------------------------------------------------------------------------
__global__ void prep_w5_k(const float* __restrict__ w0,
                          const float* __restrict__ w1,
                          const float* __restrict__ w2,
                          const float* __restrict__ w3,
                          const float* __restrict__ w4,
                          __bf16* __restrict__ Wp) {
  const int m = blockIdx.y;
  const float* W = (m == 0) ? w0 : (m == 1) ? w1 : (m == 2) ? w2
                   : (m == 3) ? w3 : w4;
  size_t idx = (size_t)blockIdx.x * blockDim.x + threadIdx.x;  // < 2M
  int t    = (int)(idx >> 9);
  int r    = (int)(idx & 511);
  int lane = r >> 4;
  int i    = r & 15;
  int nt   = t >> 6;
  int kt   = t & 63;
  int k = kt * 32 + ((lane >> 4) << 4) + i;
  int n = nt * 16 + (lane & 15);
  Wp[(size_t)m * KDIM * DIM + idx] = (__bf16)W[(size_t)k * DIM + n];
}

// ---------------------------------------------------------------------------
// Init per-call state: z = 0, logit = 0, mkey = 0 (== -inf key), den = 0
// ---------------------------------------------------------------------------
__global__ void init_k(unsigned* __restrict__ mkey, float* __restrict__ den,
                       float* __restrict__ z, float* __restrict__ logit) {
  size_t i = (size_t)blockIdx.x * blockDim.x + threadIdx.x;  // < N*D
  z[i] = 0.f;
  if (i < NEDGE) logit[i] = 0.f;
  if (i < NNODE) { mkey[i] = 0u; den[i] = 0.f; }
}

// ---------------------------------------------------------------------------
// Fused edge MLP + attention logit (partial over a 512-col half).
// Block: 512 threads = 16 waves; tile = 64 edges x 512 cols, K = 2048.
// blockIdx.y = column half. Wave (mt = w&3, ng = w>>2): M-tile mt,
// n-tiles [8*ng, 8*ng+8) within the half.  Partial logits are atomically
// added into logit[] (zero-initialized).  b_a is dropped: softmax is
// invariant under a constant shift of all logits in a segment.
// B k-panels (32KB) double-buffered in LDS, staged by TDM / async-to-LDS.
// ---------------------------------------------------------------------------
__global__ __launch_bounds__(512, 1)
void edge_gemm_logit_k(const __bf16* __restrict__ Anode,
                       const __bf16* __restrict__ Wpall,
                       const float* __restrict__ bhh,
                       const float* __restrict__ boo,
                       const float* __restrict__ bho,
                       const float* __restrict__ Wa,
                       const int* __restrict__ esrc,
                       const int* __restrict__ edst,
                       float* __restrict__ logit) {
  __shared__ __bf16 Bsh[2][32 * 512];   // 2 x 32KB double-buffered B panel
  __shared__ float logacc[64];

  const int tid  = threadIdx.x;
  const int lane = tid & 31;
  const int wave = tid >> 5;
  const int mt   = wave & 3;
  const int ng   = wave >> 2;
  const int eblk = blockIdx.x * 64;
  const int cb   = blockIdx.y;          // column half: cols [cb*512, cb*512+512)

  const int etype = (eblk < EHH) ? 0 : (eblk < EHH + EOO ? 1 : 2);
  // Panel base for this column half: n-tiles [cb*32, cb*32+32)
  const __bf16* Wcb = Wpall + (size_t)etype * KDIM * DIM
                      + (size_t)cb * (32 * 64 * 512);
  const float* bias = (etype == 0) ? bhh : (etype == 1) ? boo : bho;

  const unsigned ldsbase = (unsigned)(uintptr_t)(&Bsh[0][0]);

  const int erow = eblk + mt * 16 + (lane & 15);
  const __bf16* rowS = Anode + (size_t)esrc[erow] * KDIM;
  const __bf16* rowD = Anode + (size_t)edst[erow] * KDIM;
  const int sub = (lane >> 4) * 16;

  v8f acc[8] = {};

  if (tid < 64) logacc[tid] = 0.f;

  // Prologue: stage panel 0 into buffer 0.
  stage_panel(Wcb, 0, ldsbase, tid, wave);
  stage_wait();
  __syncthreads();

  for (int ks = 0; ks < 64; ++ks) {
    const int cur = ks & 1;
    if (ks + 1 < 64)  // prefetch next panel into the other buffer (async)
      stage_panel(Wcb, ks + 1, ldsbase + (unsigned)(cur ^ 1) * 32768u, tid, wave);

    // Gathered A fragment: cols [0,1024) from src row, [1024,2048) from dst.
    const __bf16* rp = ((ks < 32) ? rowS : rowD) + ((ks & 31) * 32);
    union { v16bf v; v8bf h[2]; } af;
    af.h[0] = *(const v8bf*)(rp + sub);
    af.h[1] = *(const v8bf*)(rp + sub + 8);

    const __bf16* Bcur = &Bsh[cur][0];
#pragma unroll
    for (int j = 0; j < 8; ++j) {
      const int nt = ng * 8 + j;      // local n-tile within the half
      v16bf bf = *(const v16bf*)(Bcur + nt * 512 + lane * 16);
      acc[j] = __builtin_amdgcn_wmma_f32_16x16x32_bf16(
          false, af.v, false, bf, (short)0, acc[j], false, false);
    }

    stage_wait();        // my staged panel landed in LDS
    __syncthreads();     // everyone done reading cur, next panel visible
  }

  // Epilogue: relu(acc + b) dotted with W_a, reduced to per-edge partial logit.
  float p[8] = {0, 0, 0, 0, 0, 0, 0, 0};
#pragma unroll
  for (int j = 0; j < 8; ++j) {
    const int col  = (cb * 32 + ng * 8 + j) * 16 + (lane & 15);
    const float w  = Wa[col];
    const float bc = bias[col];
#pragma unroll
    for (int v = 0; v < 8; ++v) {
      float x = acc[j][v] + bc;
      x = x > 0.f ? x : 0.f;
      p[v] += x * w;
    }
  }
#pragma unroll
  for (int off = 1; off < 16; off <<= 1)
#pragma unroll
    for (int v = 0; v < 8; ++v) p[v] += __shfl_xor(p[v], off, 32);

  if ((lane & 15) == 0) {
    const int rbase = mt * 16 + (lane >> 4) * 8;  // D rows: 8*(l>>4)+v
#pragma unroll
    for (int v = 0; v < 8; ++v) atomicAdd(&logacc[rbase + v], p[v]);
  }
  __syncthreads();
  if (tid < 64) atomicAdd(&logit[eblk + tid], logacc[tid]);
}

// ---------------------------------------------------------------------------
// Segment max over dst, then ex = exp(logit - m[dst]) and den = segsum(ex)
// ---------------------------------------------------------------------------
__global__ void seg_max_k(const float* __restrict__ logit,
                          const int* __restrict__ edst,
                          unsigned* __restrict__ mkey) {
  int e = blockIdx.x * blockDim.x + threadIdx.x;
  atomicMax(&mkey[edst[e]], ordkey(logit[e]));
}

__global__ void exden_k(float* __restrict__ logit,  // in: logit, out: ex
                        const int* __restrict__ edst,
                        const unsigned* __restrict__ mkey,
                        float* __restrict__ den) {
  int e = blockIdx.x * blockDim.x + threadIdx.x;
  int d = edst[e];
  float ex = __expf(logit[e] - orddec(mkey[d]));
  logit[e] = ex;
  atomicAdd(&den[d], ex);
}

// ---------------------------------------------------------------------------
// z[dst] += (ex/den[dst]) * nf[src]   (one block per edge, f32 atomics to L2)
// ---------------------------------------------------------------------------
__global__ __launch_bounds__(256)
void aggregate_k(const float* __restrict__ nf, const float* __restrict__ ex,
                 const float* __restrict__ den, const int* __restrict__ esrc,
                 const int* __restrict__ edst, float* __restrict__ z) {
  const int e = blockIdx.x;
  const int d = edst[e];
  const float alpha = ex[e] / den[d];
  const float4 v =
      ((const float4*)(nf + (size_t)esrc[e] * DIM))[threadIdx.x];
  float* zr = z + (size_t)d * DIM + threadIdx.x * 4;
  atomicAdd(zr + 0, alpha * v.x);
  atomicAdd(zr + 1, alpha * v.y);
  atomicAdd(zr + 2, alpha * v.z);
  atomicAdd(zr + 3, alpha * v.w);
}

// z (f32) -> bf16 into A_node[:, 1024:2048]
__global__ void z2bf_k(const float* __restrict__ z,
                       __bf16* __restrict__ Anode) {
  size_t i = (size_t)blockIdx.x * blockDim.x + threadIdx.x;  // < N*D
  size_t row = i >> 10;
  size_t col = i & 1023;
  Anode[row * KDIM + 1024 + col] = (__bf16)z[i];
}

// ---------------------------------------------------------------------------
// Node MLP: out = relu(A_node @ W_{hn|on} + b)
// Same tiling as the edge GEMM: 512 threads, 64 rows x 512 cols per block,
// blockIdx.y = column half (disjoint output columns, direct stores).
// ---------------------------------------------------------------------------
__global__ __launch_bounds__(512, 1)
void node_gemm_k(const __bf16* __restrict__ Anode,
                 const __bf16* __restrict__ Wpnode,  // [hn | on] packed
                 const float* __restrict__ bhn,
                 const float* __restrict__ bon,
                 float* __restrict__ out) {
  __shared__ __bf16 Bsh[2][32 * 512];

  const int tid  = threadIdx.x;
  const int lane = tid & 31;
  const int wave = tid >> 5;
  const int mt   = wave & 3;
  const int ng   = wave >> 2;
  const int rblk = blockIdx.x * 64;
  const int cb   = blockIdx.y;

  const int ntype = (rblk < NHYD) ? 0 : 1;
  const __bf16* Wcb = Wpnode + (size_t)ntype * KDIM * DIM
                      + (size_t)cb * (32 * 64 * 512);
  const float* bias = ntype ? bon : bhn;

  const unsigned ldsbase = (unsigned)(uintptr_t)(&Bsh[0][0]);

  const __bf16* rowp =
      Anode + (size_t)(rblk + mt * 16 + (lane & 15)) * KDIM;
  const int sub = (lane >> 4) * 16;

  v8f acc[8] = {};

  stage_panel(Wcb, 0, ldsbase, tid, wave);
  stage_wait();
  __syncthreads();

  for (int ks = 0; ks < 64; ++ks) {
    const int cur = ks & 1;
    if (ks + 1 < 64)
      stage_panel(Wcb, ks + 1, ldsbase + (unsigned)(cur ^ 1) * 32768u, tid, wave);

    const __bf16* rp = rowp + ks * 32;
    union { v16bf v; v8bf h[2]; } af;
    af.h[0] = *(const v8bf*)(rp + sub);
    af.h[1] = *(const v8bf*)(rp + sub + 8);

    const __bf16* Bcur = &Bsh[cur][0];
#pragma unroll
    for (int j = 0; j < 8; ++j) {
      v16bf bf = *(const v16bf*)(Bcur + (ng * 8 + j) * 512 + lane * 16);
      acc[j] = __builtin_amdgcn_wmma_f32_16x16x32_bf16(
          false, af.v, false, bf, (short)0, acc[j], false, false);
    }

    stage_wait();
    __syncthreads();
  }

  const int rowOut = rblk + mt * 16 + (lane >> 4) * 8;
#pragma unroll
  for (int j = 0; j < 8; ++j) {
    const int col  = (cb * 32 + ng * 8 + j) * 16 + (lane & 15);
    const float bc = bias[col];
#pragma unroll
    for (int v = 0; v < 8; ++v) {
      float x = acc[j][v] + bc;
      out[(size_t)(rowOut + v) * DIM + col] = x > 0.f ? x : 0.f;
    }
  }
}

// ---------------------------------------------------------------------------
// Host-side orchestration
// ---------------------------------------------------------------------------
extern "C" void kernel_launch(void* const* d_in, const int* in_sizes, int n_in,
                              void* d_out, int out_size, void* d_ws,
                              size_t ws_size, hipStream_t stream) {
  (void)in_sizes; (void)n_in; (void)out_size; (void)ws_size;
  const float* nf  = (const float*)d_in[0];
  const float* Whh = (const float*)d_in[1];
  const float* bhh = (const float*)d_in[2];
  const float* Woo = (const float*)d_in[3];
  const float* boo = (const float*)d_in[4];
  const float* Who = (const float*)d_in[5];
  const float* bho = (const float*)d_in[6];
  const float* Wa  = (const float*)d_in[7];
  const float* Whn = (const float*)d_in[9];
  const float* bhn = (const float*)d_in[10];
  const float* Won = (const float*)d_in[11];
  const float* bon = (const float*)d_in[12];
  const int* esrc  = (const int*)d_in[13];
  const int* edst  = (const int*)d_in[14];
  float* out = (float*)d_out;

  char* ws = (char*)d_ws;
  size_t off = 0;
  __bf16* Anode = (__bf16*)(ws + off); off += (size_t)NNODE * KDIM * 2;      // 64 MB
  __bf16* Wp    = (__bf16*)(ws + off); off += 5ull * KDIM * DIM * 2;         // 20 MB
  float* logit  = (float*)(ws + off);  off += (size_t)NEDGE * 4;             // 256 KB
  unsigned* mkey = (unsigned*)(ws + off); off += (size_t)NNODE * 4;          // 64 KB
  float* den    = (float*)(ws + off);  off += (size_t)NNODE * 4;             // 64 KB
  float* z      = (float*)(ws + off);  off += (size_t)NNODE * DIM * 4;       // 64 MB

  const int NELEM_BLKS = (NNODE * DIM) / 256;  // 65536

  prep_nf_k<<<NELEM_BLKS, 256, 0, stream>>>(nf, Anode);
  {
    dim3 g((KDIM * DIM) / 256, 5);
    prep_w5_k<<<g, 256, 0, stream>>>(Whh, Woo, Who, Whn, Won, Wp);
  }
  init_k<<<NELEM_BLKS, 256, 0, stream>>>(mkey, den, z, logit);

  {
    dim3 g(NEDGE / 64, 2);   // 1024 row-blocks x 2 column halves
    edge_gemm_logit_k<<<g, 512, 0, stream>>>(
        Anode, Wp, bhh, boo, bho, Wa, esrc, edst, logit);
  }

  seg_max_k<<<NEDGE / 256, 256, 0, stream>>>(logit, edst, mkey);
  exden_k<<<NEDGE / 256, 256, 0, stream>>>(logit, edst, mkey, den);
  aggregate_k<<<NEDGE, 256, 0, stream>>>(nf, logit, den, esrc, edst, z);
  z2bf_k<<<NELEM_BLKS, 256, 0, stream>>>(z, Anode);

  {
    dim3 g(NNODE / 64, 2);   // 256 row-blocks x 2 column halves
    node_gemm_k<<<g, 512, 0, stream>>>(
        Anode, Wp + 3ull * KDIM * DIM, bhn, bon, out);
  }
}
